// StatefulCausalHopfieldAttentionHead_34342558498821
// MI455X (gfx1250) — matrix operand order repaired and checked by reading, not compiled, not run on hardware
//
#include <hip/hip_runtime.h>

#define T_TOT 2304
#define D_H   64
#define D_IN  1024
#define NBATCH 2
#define S_LEN 128
#define NEGV  (-1.0e30f)
#define NBLK  (NBATCH * (T_TOT / 16))   // 288

typedef __attribute__((ext_vector_type(16))) __bf16 bf16x16;
typedef __attribute__((ext_vector_type(8)))  __bf16 bf16x8;
typedef __attribute__((ext_vector_type(8)))  float  f32x8;

static __device__ inline bf16x16 make16(bf16x8 lo, bf16x8 hi) {
    union { bf16x16 v; bf16x8 h[2]; } u;
    u.h[0] = lo; u.h[1] = hi;
    return u.v;
}

// A-fragment (16x32 bf16, row-major source, lda elements):
// lane: m = lane&15, half = lane>>4 ; element i -> k = (i/8)*16 + half*8 + (i%8)
static __device__ inline bf16x16 load_a(const __bf16* base, int lda, int k0, int lane) {
    int m = lane & 15, half = lane >> 4;
    const __bf16* p = base + (size_t)m * lda + k0 + half * 8;
    bf16x8 lo = *(const bf16x8*)p;
    bf16x8 hi = *(const bf16x8*)(p + 16);
    return make16(lo, hi);
}

// B-fragment (32x16 bf16) from K-contiguous ("transposed") source: elem (k,n) at baseT[n*ldb + k]
// lane: n = lane&15, half = lane>>4 ; element i -> k = (i/8)*16 + half*8 + (i%8)
static __device__ inline bf16x16 load_b(const __bf16* baseT, int ldb, int k0, int lane) {
    int n = lane & 15, half = lane >> 4;
    const __bf16* p = baseT + (size_t)n * ldb + k0 + half * 8;
    bf16x8 lo = *(const bf16x8*)p;
    bf16x8 hi = *(const bf16x8*)(p + 16);
    return make16(lo, hi);
}

static __device__ inline f32x8 wmma_bf16(bf16x16 a, bf16x16 b, f32x8 c) {
    return __builtin_amdgcn_wmma_f32_16x16x32_bf16(false, a, false, b, (short)0, c, false, false);
}

static __device__ inline float wave_sum(float v) {
    for (int o = 16; o > 0; o >>= 1) v += __shfl_xor(v, o, 32);
    return v;
}
static __device__ inline float wave_max(float v) {
    for (int o = 16; o > 0; o >>= 1) v = fmaxf(v, __shfl_xor(v, o, 32));
    return v;
}

// ---------------- prep: f32 (I,O) -> bf16 transposed (O,I) ----------------
__global__ void k_transpose_bf16(const float* __restrict__ in, __bf16* __restrict__ out,
                                 int I, int O) {
    int t = blockIdx.x * 256 + threadIdx.x;
    if (t >= I * O) return;
    int i = t / O, o = t % O;
    out[(size_t)o * I + i] = (__bf16)in[t];
}

// ---------------- K1: LayerNorm + projection to mem (bf16) ----------------
__global__ void k_ln_proj(const float* __restrict__ x,
                          const __bf16* __restrict__ WhT,   // mid   (64,1024)
                          const __bf16* __restrict__ WhsT,  // start
                          const __bf16* __restrict__ WheT,  // end
                          const float* __restrict__ ln_g,  const float* __restrict__ ln_b,
                          const float* __restrict__ ln_gs, const float* __restrict__ ln_bs,
                          const float* __restrict__ ln_ge, const float* __restrict__ ln_be,
                          __bf16* __restrict__ memA) {
    __shared__ __bf16 xn[16 * D_IN];   // 32 KB
    __shared__ float  Obuf[16 * D_H];  // 4 KB

    int tid = threadIdx.x, lane = tid & 31, w = tid >> 5;
    int blk = blockIdx.x;
    int b = blk / (T_TOT / 16);
    int t0 = (blk % (T_TOT / 16)) * 16;

    int seg = (t0 < S_LEN) ? 0 : ((t0 >= T_TOT - S_LEN) ? 2 : 1);
    const float* g  = (seg == 0) ? ln_gs : (seg == 2) ? ln_ge : ln_g;
    const float* bb = (seg == 0) ? ln_bs : (seg == 2) ? ln_be : ln_b;
    const __bf16* Wt = (seg == 0) ? WhsT : (seg == 2) ? WheT : WhT;

    for (int i = tid; i < 16 * D_H; i += 256) Obuf[i] = 0.f;

    // LayerNorm: each wave handles 2 rows
    for (int rr = 0; rr < 2; ++rr) {
        int m = w * 2 + rr;
        const float* xr = x + (size_t)(b * T_TOT + t0 + m) * D_IN;
        float s = 0.f, s2 = 0.f;
        for (int i = lane; i < D_IN; i += 32) { float v = xr[i]; s += v; s2 += v * v; }
        s = wave_sum(s); s2 = wave_sum(s2);
        float mean = s * (1.f / D_IN);
        float var  = s2 * (1.f / D_IN) - mean * mean;
        float rstd = rsqrtf(var + 1e-5f);
        for (int i = lane; i < D_IN; i += 32) {
            float v = (xr[i] - mean) * rstd * g[i] + bb[i];
            xn[m * D_IN + i] = (__bf16)v;
        }
    }
    __syncthreads();

    // GEMM 16x1024 @ 1024x64: wave = (ntile, khalf)
    int nt = w & 3, kh = w >> 2;
    f32x8 acc = {0.f, 0.f, 0.f, 0.f, 0.f, 0.f, 0.f, 0.f};
    for (int c = kh * 16; c < kh * 16 + 16; ++c) {
        int k0 = c * 32;
        bf16x16 a = load_a(xn, D_IN, k0, lane);
        bf16x16 bfr = load_b(Wt + (size_t)nt * 16 * D_IN, D_IN, k0, lane);
        acc = wmma_bf16(a, bfr, acc);
    }
    {
        int n = lane & 15, half = lane >> 4;
        #pragma unroll
        for (int r = 0; r < 8; ++r)
            atomicAdd(&Obuf[(r + 8 * half) * D_H + nt * 16 + n], acc[r]);
    }
    __syncthreads();

    for (int e = tid; e < 16 * D_H; e += 256) {
        int m = e >> 6, d = e & 63;
        memA[(size_t)(b * T_TOT + t0 + m) * D_H + d] = (__bf16)Obuf[e];
    }
}

// ---------------- K2: qkv projections (per-segment 64x64 weights) ----------------
__global__ void k_qkv(const __bf16* __restrict__ mem,
                      const __bf16* __restrict__ WTq,  const __bf16* __restrict__ WTk,  const __bf16* __restrict__ WTv,
                      const __bf16* __restrict__ WTqs, const __bf16* __restrict__ WTks, const __bf16* __restrict__ WTvs,
                      const __bf16* __restrict__ WTqe, const __bf16* __restrict__ WTke, const __bf16* __restrict__ WTve,
                      __bf16* __restrict__ qo, __bf16* __restrict__ ko, __bf16* __restrict__ vT) {
    int tid = threadIdx.x, lane = tid & 31, w = tid >> 5;
    int blk = blockIdx.x;
    int b = blk / (T_TOT / 16);
    int t0 = (blk % (T_TOT / 16)) * 16;
    int seg = (t0 < S_LEN) ? 0 : ((t0 >= T_TOT - S_LEN) ? 2 : 1);

    const __bf16* Wq = (seg == 0) ? WTqs : (seg == 2) ? WTqe : WTq;
    const __bf16* Wk = (seg == 0) ? WTks : (seg == 2) ? WTke : WTk;
    const __bf16* Wv = (seg == 0) ? WTvs : (seg == 2) ? WTve : WTv;

    const __bf16* arow = mem + (size_t)(b * T_TOT + t0) * D_H;
    // A-fragments are identical for every task: hoist
    bf16x16 a0 = load_a(arow, D_H, 0, lane);
    bf16x16 a1 = load_a(arow, D_H, 32, lane);

    for (int task = w; task < 12; task += 8) {
        int out_id = task >> 2, nt = task & 3;
        const __bf16* Wt = (out_id == 0) ? Wq : (out_id == 1) ? Wk : Wv;
        f32x8 acc = {0.f, 0.f, 0.f, 0.f, 0.f, 0.f, 0.f, 0.f};
        acc = wmma_bf16(a0, load_b(Wt + (size_t)nt * 16 * D_H, D_H, 0, lane), acc);
        acc = wmma_bf16(a1, load_b(Wt + (size_t)nt * 16 * D_H, D_H, 32, lane), acc);
        int n = lane & 15, half = lane >> 4;
        if (out_id == 2) {
            // v stored transposed: vT[(b*64 + d)*T + s]
            #pragma unroll
            for (int r = 0; r < 8; ++r)
                vT[((size_t)b * D_H + nt * 16 + n) * T_TOT + t0 + r + 8 * half] = (__bf16)acc[r];
        } else {
            __bf16* o = (out_id == 0) ? qo : ko;
            #pragma unroll
            for (int r = 0; r < 8; ++r)
                o[(size_t)(b * T_TOT + t0 + r + 8 * half) * D_H + nt * 16 + n] = (__bf16)acc[r];
        }
    }
}

// ---------------- K3: fused attention row-block (16 q rows per WG) ----------------
// LDS: Ls[16][2304] logits->scores->bf16 probs ; LI[16][2304] cope logits (iter0) ; Obuf[16][64]
__global__ void k_attn(const __bf16* __restrict__ q, const __bf16* __restrict__ k,
                       const __bf16* __restrict__ vT, const __bf16* __restrict__ copeT,
                       float* __restrict__ outF, __bf16* __restrict__ outB, int use_cope) {
    extern __shared__ float smem[];
    float* Ls = smem;
    float* LI = use_cope ? (smem + 16 * T_TOT) : nullptr;
    float* Obuf = smem + (size_t)16 * T_TOT * (use_cope ? 2 : 1);

    int tid = threadIdx.x, lane = tid & 31, w = tid >> 5;
    int blk = blockIdx.x;
    int b = blk / (T_TOT / 16);
    int t0 = (blk % (T_TOT / 16)) * 16;

    const __bf16* qrow = q + (size_t)(b * T_TOT + t0) * D_H;
    const __bf16* kb   = k + (size_t)b * T_TOT * D_H;
    const __bf16* vTb  = vT + (size_t)b * D_H * T_TOT;

    // q A-fragments are invariant over all s-tiles: load once
    bf16x16 qa0 = load_a(qrow, D_H, 0, lane);
    bf16x16 qa1 = load_a(qrow, D_H, 32, lane);

    // ---- Phase 1: logits (and cope logits) via WMMA, 144 s-tiles over 8 waves ----
    for (int st = w; st < T_TOT / 16; st += 8) {
        if (st + 8 < T_TOT / 16) {   // prefetch next tile one iteration ahead
            __builtin_prefetch(kb + (size_t)(st + 8) * 16 * D_H + lane * 32, 0, 0);
            if (use_cope)
                __builtin_prefetch(copeT + (size_t)(st + 8) * 16 * D_H + lane * 32, 0, 0);
        }
        f32x8 acc  = {0.f, 0.f, 0.f, 0.f, 0.f, 0.f, 0.f, 0.f};
        f32x8 acc2 = {0.f, 0.f, 0.f, 0.f, 0.f, 0.f, 0.f, 0.f};
        acc = wmma_bf16(qa0, load_b(kb + (size_t)st * 16 * D_H, D_H, 0, lane), acc);
        acc = wmma_bf16(qa1, load_b(kb + (size_t)st * 16 * D_H, D_H, 32, lane), acc);
        if (use_cope) {
            acc2 = wmma_bf16(qa0, load_b(copeT + (size_t)st * 16 * D_H, D_H, 0, lane), acc2);
            acc2 = wmma_bf16(qa1, load_b(copeT + (size_t)st * 16 * D_H, D_H, 32, lane), acc2);
        }
        int n = lane & 15, half = lane >> 4;
        int scol = st * 16 + n;
        #pragma unroll
        for (int r = 0; r < 8; ++r) {
            int m = r + 8 * half;
            Ls[m * T_TOT + scol] = acc[r];
            if (use_cope) LI[m * T_TOT + scol] = acc2[r];
        }
    }
    for (int i = tid; i < 16 * D_H; i += 256) Obuf[i] = 0.f;
    __syncthreads();

    // ---- Phase 2: CoPE bias + mask + softmax, 2 rows per wave ----
    const float scale = 0.125f; // 1/sqrt(64)
    for (int rr = 0; rr < 2; ++rr) {
        int m = w * 2 + rr;
        int tg = t0 + m;
        float* row = Ls + m * T_TOT;
        float mx = -3.4e38f;
        if (use_cope) {
            const float* li = LI + m * T_TOT;
            float carry = 0.f;
            for (int c = T_TOT / 32 - 1; c >= 0; --c) {   // reversed cumsum: high chunks first
                int idx = c * 32 + lane;
                float lg = row[idx];
                float gte = 1.f / (1.f + __expf(-lg));
                float r = gte;
                #pragma unroll
                for (int off = 1; off < 32; off <<= 1) {  // suffix-sum within chunk
                    float y = __shfl_down(r, off, 32);
                    if (lane + off < 32) r += y;
                }
                float pos = fminf(r + carry, (float)(T_TOT - 1));
                float pf = floorf(pos), pc = ceilf(pos);
                float wf = pos - pf;
                float lc = li[(int)pc];
                float lf = li[(int)pf];
                float bias = lc * wf + lf * (1.f - wf);
                float sc = lg * scale + ((idx <= tg) ? 0.f : NEGV) + bias;
                row[idx] = sc;
                mx = fmaxf(mx, sc);
                carry += __shfl(r, 0, 32);                // chunk total
            }
        } else {
            for (int c = 0; c < T_TOT / 32; ++c) {
                int idx = c * 32 + lane;
                float sc = row[idx] * scale + ((idx <= tg) ? 0.f : NEGV);
                row[idx] = sc;
                mx = fmaxf(mx, sc);
            }
        }
        mx = wave_max(mx);
        float sum = 0.f;
        for (int c = 0; c < T_TOT / 32; ++c) {
            int idx = c * 32 + lane;
            float e = __expf(row[idx] - mx);
            row[idx] = e;
            sum += e;
        }
        sum = wave_sum(sum);
        float inv = 1.f / sum;
        // Normalize and pack probabilities as bf16 into the LOW half of the same row.
        // Packed write at float index idx/2 never passes the float read cursor idx.
        __bf16* pb = (__bf16*)row;
        for (int c = 0; c < T_TOT / 32; ++c) {
            int idx = c * 32 + lane;
            float pv = row[idx] * inv;
            pb[idx] = (__bf16)pv;
        }
    }
    __syncthreads();

    // ---- Phase 3: out = attn @ v, K=2304 split over 8 waves, 4 n-tiles each ----
    // A-fragments now come straight from the bf16-packed rows: 2 x ds_load_b128.
    const __bf16* probs = (const __bf16*)Ls;     // row stride = 2*T_TOT bf16 elements
    f32x8 acc[4];
    #pragma unroll
    for (int nt = 0; nt < 4; ++nt) acc[nt] = (f32x8){0.f, 0.f, 0.f, 0.f, 0.f, 0.f, 0.f, 0.f};

    for (int c = w; c < T_TOT / 32; c += 8) {
        if (c + 8 < T_TOT / 32) {   // prefetch next v chunk (rows d = 2*lane, 2*lane+1)
            __builtin_prefetch(vTb + (size_t)(lane * 2)     * T_TOT + (c + 8) * 32, 0, 0);
            __builtin_prefetch(vTb + (size_t)(lane * 2 + 1) * T_TOT + (c + 8) * 32, 0, 0);
        }
        bf16x16 a = load_a(probs, 2 * T_TOT, c * 32, lane);
        #pragma unroll
        for (int nt = 0; nt < 4; ++nt) {
            bf16x16 bv = load_b(vTb + (size_t)nt * 16 * T_TOT, T_TOT, c * 32, lane);
            acc[nt] = wmma_bf16(a, bv, acc[nt]);
        }
    }
    {
        int n = lane & 15, half = lane >> 4;
        #pragma unroll
        for (int nt = 0; nt < 4; ++nt)
            #pragma unroll
            for (int r = 0; r < 8; ++r)
                atomicAdd(&Obuf[(r + 8 * half) * D_H + nt * 16 + n], acc[nt][r]);
    }
    __syncthreads();

    for (int e = tid; e < 16 * D_H; e += 256) {
        int m = e >> 6, d = e & 63;
        float val = Obuf[e];
        size_t o = (size_t)(b * T_TOT + t0 + m) * D_H + d;
        if (outF) outF[o] = val;
        if (outB) outB[o] = (__bf16)val;
    }
}

// ---------------- host ----------------
extern "C" void kernel_launch(void* const* d_in, const int* in_sizes, int n_in,
                              void* d_out, int out_size, void* d_ws, size_t ws_size,
                              hipStream_t stream) {
    const float* x     = (const float*)d_in[0];
    const float* Wh    = (const float*)d_in[1];
    const float* Whs   = (const float*)d_in[2];
    const float* Whe   = (const float*)d_in[3];
    const float* Wk    = (const float*)d_in[4];
    const float* Wq    = (const float*)d_in[5];
    const float* Wv    = (const float*)d_in[6];
    const float* Wks   = (const float*)d_in[7];
    const float* Wqs   = (const float*)d_in[8];
    const float* Wvs   = (const float*)d_in[9];
    const float* Wke   = (const float*)d_in[10];
    const float* Wqe   = (const float*)d_in[11];
    const float* Wve   = (const float*)d_in[12];
    const float* ln_g  = (const float*)d_in[13];
    const float* ln_b  = (const float*)d_in[14];
    const float* ln_gs = (const float*)d_in[15];
    const float* ln_bs = (const float*)d_in[16];
    const float* ln_ge = (const float*)d_in[17];
    const float* ln_be = (const float*)d_in[18];
    const float* cope  = (const float*)d_in[19];

    char* base = (char*)d_ws;
    size_t off = 0;
    auto alloc = [&](size_t bytes) -> char* {
        char* p = base + off;
        off = (off + bytes + 255) & ~(size_t)255;
        return p;
    };

    __bf16* WhT  = (__bf16*)alloc((size_t)D_H * D_IN * 2);
    __bf16* WhsT = (__bf16*)alloc((size_t)D_H * D_IN * 2);
    __bf16* WheT = (__bf16*)alloc((size_t)D_H * D_IN * 2);
    __bf16* WTk  = (__bf16*)alloc(D_H * D_H * 2);
    __bf16* WTq  = (__bf16*)alloc(D_H * D_H * 2);
    __bf16* WTv  = (__bf16*)alloc(D_H * D_H * 2);
    __bf16* WTks = (__bf16*)alloc(D_H * D_H * 2);
    __bf16* WTqs = (__bf16*)alloc(D_H * D_H * 2);
    __bf16* WTvs = (__bf16*)alloc(D_H * D_H * 2);
    __bf16* WTke = (__bf16*)alloc(D_H * D_H * 2);
    __bf16* WTqe = (__bf16*)alloc(D_H * D_H * 2);
    __bf16* WTve = (__bf16*)alloc(D_H * D_H * 2);
    __bf16* copeT = (__bf16*)alloc((size_t)T_TOT * D_H * 2);
    __bf16* memA = (__bf16*)alloc((size_t)NBATCH * T_TOT * D_H * 2);
    __bf16* memB = (__bf16*)alloc((size_t)NBATCH * T_TOT * D_H * 2);
    __bf16* qb   = (__bf16*)alloc((size_t)NBATCH * T_TOT * D_H * 2);
    __bf16* kbuf = (__bf16*)alloc((size_t)NBATCH * T_TOT * D_H * 2);
    __bf16* vTb  = (__bf16*)alloc((size_t)NBATCH * T_TOT * D_H * 2);
    (void)ws_size; (void)in_sizes; (void)n_in; (void)out_size;

    // prep: transpose/convert weights and cope table to bf16, K-contiguous
    int nBig = D_IN * D_H;
    k_transpose_bf16<<<(nBig + 255) / 256, 256, 0, stream>>>(Wh,  WhT,  D_IN, D_H);
    k_transpose_bf16<<<(nBig + 255) / 256, 256, 0, stream>>>(Whs, WhsT, D_IN, D_H);
    k_transpose_bf16<<<(nBig + 255) / 256, 256, 0, stream>>>(Whe, WheT, D_IN, D_H);
    int nSm = D_H * D_H;
    k_transpose_bf16<<<(nSm + 255) / 256, 256, 0, stream>>>(Wk,  WTk,  D_H, D_H);
    k_transpose_bf16<<<(nSm + 255) / 256, 256, 0, stream>>>(Wq,  WTq,  D_H, D_H);
    k_transpose_bf16<<<(nSm + 255) / 256, 256, 0, stream>>>(Wv,  WTv,  D_H, D_H);
    k_transpose_bf16<<<(nSm + 255) / 256, 256, 0, stream>>>(Wks, WTks, D_H, D_H);
    k_transpose_bf16<<<(nSm + 255) / 256, 256, 0, stream>>>(Wqs, WTqs, D_H, D_H);
    k_transpose_bf16<<<(nSm + 255) / 256, 256, 0, stream>>>(Wvs, WTvs, D_H, D_H);
    k_transpose_bf16<<<(nSm + 255) / 256, 256, 0, stream>>>(Wke, WTke, D_H, D_H);
    k_transpose_bf16<<<(nSm + 255) / 256, 256, 0, stream>>>(Wqe, WTqe, D_H, D_H);
    k_transpose_bf16<<<(nSm + 255) / 256, 256, 0, stream>>>(Wve, WTve, D_H, D_H);
    int nCp = D_H * T_TOT;
    k_transpose_bf16<<<(nCp + 255) / 256, 256, 0, stream>>>(cope, copeT, D_H, T_TOT);

    // K1: LN + projection -> memA
    k_ln_proj<<<NBLK, 256, 0, stream>>>(x, WhT, WhsT, WheT,
                                        ln_g, ln_b, ln_gs, ln_bs, ln_ge, ln_be, memA);

    // iter 0
    k_qkv<<<NBLK, 256, 0, stream>>>(memA, WTq, WTk, WTv, WTqs, WTks, WTvs, WTqe, WTke, WTve,
                                    qb, kbuf, vTb);
    size_t shA = ((size_t)16 * T_TOT * 2 + 16 * D_H) * sizeof(float);  // ~299 KB
    k_attn<<<NBLK, 256, shA, stream>>>(qb, kbuf, vTb, copeT, nullptr, memB, 1);

    // iter 1
    k_qkv<<<NBLK, 256, 0, stream>>>(memB, WTq, WTk, WTv, WTqs, WTks, WTvs, WTqe, WTke, WTve,
                                    qb, kbuf, vTb);
    size_t shB = ((size_t)16 * T_TOT + 16 * D_H) * sizeof(float);      // ~148 KB
    k_attn<<<NBLK, 256, shB, stream>>>(qb, kbuf, vTb, copeT, (float*)d_out, nullptr, 0);
}
